// SecondOrderNeuralODE_84164179132484
// MI455X (gfx1250) — matrix operand order repaired and checked by reading, not compiled, not run on hardware
//
#include <hip/hip_runtime.h>
#include <math.h>

// ---------------------------------------------------------------------------
// Second-order Neural ODE (RK4, 10 steps) on gfx1250 WMMA.
// One persistent workgroup per 32-row batch tile; whole time loop in-kernel.
// GEMMs: v_wmma_f32_16x16x32_f16 with manual ping-pong double buffering.
// B-fragment base pointers are laundered through empty inline asm each
// iteration (blocks LICM hoist -> scratch spill) but loaded through explicit
// address_space(1) pointers so they lower to global_load_b128 (LOADcnt only),
// keeping the LDS A-fragment ds_loads on the independent DScnt domain.
// ---------------------------------------------------------------------------

typedef _Float16 h8   __attribute__((ext_vector_type(8)));
typedef _Float16 v16h __attribute__((ext_vector_type(16)));
typedef float    v8f  __attribute__((ext_vector_type(8)));

// Global-address-space views (force global_load, not flat_load).
typedef const _Float16 __attribute__((address_space(1))) gf16;
typedef const h8       __attribute__((address_space(1))) gh8;

#define DIMN     128
#define HIDN     512
#define MTILE    32          // batch rows per workgroup
#define LDA_IN   296         // halfs; padded input row stride (K padded 257->288)
#define LDH      520         // halfs; padded hidden row stride
#define NTHREADS 512         // 16 waves

#define KT1 9                // 288/32 K-tiles, layer 1
#define KT2 16               // 512/32 K-tiles, layers 2/3
#define NT12 32              // 512/16 N-tiles, layers 1/2
#define NT3  8               // 128/16 N-tiles, layer 3
#define FRAG_ELEMS 512       // 32x16 halfs per B fragment
#define W1F_ELEMS (KT1*NT12*FRAG_ELEMS)   // 147456
#define W2F_ELEMS (KT2*NT12*FRAG_ELEMS)   // 262144
#define W3F_ELEMS (KT2*NT3 *FRAG_ELEMS)   //  65536
#define WF_TOTAL  (W1F_ELEMS+W2F_ELEMS+W3F_ELEMS)  // 475136 halfs

// ------------------------------ helpers ------------------------------------

__device__ __forceinline__ v8f splat8(float x) {
  v8f r = {x, x, x, x, x, x, x, x};
  return r;
}

__device__ __forceinline__ v16h make16(h8 lo, h8 hi) {
  return __builtin_shufflevector(lo, hi, 0,1,2,3,4,5,6,7,8,9,10,11,12,13,14,15);
}

// Launder a global pointer so the optimizer cannot prove loads through it are
// loop-invariant (blocks LICM hoist -> scratch spill of weight fragments),
// while the address_space(1) type keeps the loads on the global path.
__device__ __forceinline__ gf16* opaq(gf16* p) {
  asm volatile("" : "+v"(p));
  return p;
}

// Branch-free tanh: 1 - 2/(e^{2x}+1).  v_exp + v_rcp, no EXEC divergence.
__device__ __forceinline__ float fast_tanh(float x) {
  float e = __expf(2.0f * x);                       // v_mul + v_exp_f32
  return 1.0f - 2.0f * __builtin_amdgcn_rcpf(e + 1.0f);
}

// A fragment (16x32 f16, row-major source in LDS) -> two ds_load_b128.
// Lane l<16: row M=l, K = kbase + {0..15};
// lane l>=16: row M=l-16, K = kbase + {16..31}.
__device__ __forceinline__ v16h load_a(const _Float16* base, int ld, int m0,
                                       int kbase, int lane) {
  const _Float16* p = base + (size_t)(m0 + (lane & 15)) * ld + kbase + ((lane >> 4) << 4);
  h8 lo = *(const h8*)p;
  h8 hi = *(const h8*)(p + 8);
  return make16(lo, hi);
}

// B fragment (32x16 f16) pre-swizzled: 32B contiguous per lane -> 2x
// global_load_b128 through the address_space(1) pointer.
__device__ __forceinline__ v16h load_b(gf16* frag, int lane) {
  gh8* p = (gh8*)(frag + (lane << 4));
  h8 lo = p[0];
  h8 hi = p[1];
  return make16(lo, hi);
}

__device__ __forceinline__ v8f wmma_f16(v16h a, v16h b, v8f c) {
  return __builtin_amdgcn_wmma_f32_16x16x32_f16(
      /*neg_a=*/false, a, /*neg_b=*/false, b,
      /*c_mod=*/(short)0, c, /*reuse_a=*/false, /*reuse_b=*/false);
}

// One hidden layer slice for one wave: 4 adjacent 16x16 output tiles
// (mt = wave>>3 selects M half; ntb = (wave&7)*4 the N-tile group).
// Manual ping-pong: two static buffer sets, two K-tiles per rolled iteration.
template<int NKT>
__device__ __forceinline__ void mlp_layer(const _Float16* A, int lda,
                                          gf16* WF,
                                          const float* bias,
                                          _Float16* H, int ldh,
                                          int wave, int lane) {
  const int mt = wave >> 3, ntb = (wave & 7) << 2, m0 = mt << 4;
  const int col = lane & 15, hi = lane >> 4;
  v8f acc[4];
#pragma unroll
  for (int j = 0; j < 4; ++j) acc[j] = splat8(bias[(ntb + j) * 16 + col]);

  v16h a0, a1, b0[4], b1[4];
  a0 = load_a(A, lda, m0, 0, lane);
  {
    gf16* fb = opaq(WF + ((size_t)ntb << 9));
#pragma unroll
    for (int j = 0; j < 4; ++j) b0[j] = load_b(fb + ((size_t)j << 9), lane);
  }

#pragma unroll 1
  for (int kk = 0; kk < NKT / 2; ++kk) {
    const int k1 = 2 * kk + 1;
    a1 = load_a(A, lda, m0, k1 << 5, lane);
    {
      gf16* fb = opaq(WF + ((size_t)(k1 * NT12 + ntb) << 9));
#pragma unroll
      for (int j = 0; j < 4; ++j) b1[j] = load_b(fb + ((size_t)j << 9), lane);
    }
#pragma unroll
    for (int j = 0; j < 4; ++j) acc[j] = wmma_f16(a0, b0[j], acc[j]);

    const int k2 = (2 * kk + 2 < NKT) ? 2 * kk + 2 : NKT - 1;  // clamped
    a0 = load_a(A, lda, m0, k2 << 5, lane);
    {
      gf16* fb = opaq(WF + ((size_t)(k2 * NT12 + ntb) << 9));
#pragma unroll
      for (int j = 0; j < 4; ++j) b0[j] = load_b(fb + ((size_t)j << 9), lane);
    }
#pragma unroll
    for (int j = 0; j < 4; ++j) acc[j] = wmma_f16(a1, b1[j], acc[j]);
  }
  if (NKT & 1) {   // peeled last K-tile (KT1 = 9)
#pragma unroll
    for (int j = 0; j < 4; ++j) acc[j] = wmma_f16(a0, b0[j], acc[j]);
  }

#pragma unroll
  for (int j = 0; j < 4; ++j) {
    const int n0 = (ntb + j) << 4;
#pragma unroll
    for (int r = 0; r < 8; ++r)
      H[(size_t)(m0 + r + (hi << 3)) * ldh + n0 + col] = (_Float16)fast_tanh(acc[j][r]);
  }
}

// ------------------------- weight pre-swizzle kernel -----------------------
// Converts fp32 weights to f16 B-fragments: for fragment (kt,nt), lane l,
// slot i -> W[kt*32 + 16*(l/16) + i][nt*16 + (l%16)], stored contiguously.
__global__ void ode_prep_kernel(const float* __restrict__ W1,
                                const float* __restrict__ W2,
                                const float* __restrict__ W3,
                                _Float16* __restrict__ wf) {
  int idx = blockIdx.x * blockDim.x + threadIdx.x;
  if (idx >= WF_TOTAL) return;
  float val;
  if (idx < W1F_ELEMS) {
    int f = idx >> 9, e = idx & 511;
    int lane = e >> 4, i = e & 15;
    int kt = f >> 5, nt = f & 31;
    int k = (kt << 5) + ((lane >> 4) << 4) + i;   // padded K in [0,288)
    int n = (nt << 4) + (lane & 15);
    val = (k < 257) ? W1[(size_t)k * HIDN + n] : 0.0f;   // rows 257..287 zero-pad
  } else if (idx < W1F_ELEMS + W2F_ELEMS) {
    int j = idx - W1F_ELEMS;
    int f = j >> 9, e = j & 511;
    int lane = e >> 4, i = e & 15;
    int kt = f >> 5, nt = f & 31;
    int k = (kt << 5) + ((lane >> 4) << 4) + i;
    int n = (nt << 4) + (lane & 15);
    val = W2[(size_t)k * HIDN + n];
  } else {
    int j = idx - W1F_ELEMS - W2F_ELEMS;
    int f = j >> 9, e = j & 511;
    int lane = e >> 4, i = e & 15;
    int kt = f >> 3, nt = f & 7;
    int k = (kt << 5) + ((lane >> 4) << 4) + i;
    int n = (nt << 4) + (lane & 15);
    val = W3[(size_t)k * DIMN + n];
  }
  wf[idx] = (_Float16)val;
}

// ------------------------------ main kernel --------------------------------
__global__ __launch_bounds__(NTHREADS, 1)
void ode_main_kernel(const float* __restrict__ x,
                     const float* __restrict__ b1,
                     const float* __restrict__ b2,
                     const float* __restrict__ b3,
                     const _Float16* __restrict__ wf,
                     float* __restrict__ out) {
  __shared__ float sb1[HIDN];
  __shared__ float sb2[HIDN];
  __shared__ float sb3[DIMN];
  __shared__ float s_x[MTILE * 256];                    // current state [x|v], f32
  __shared__ float s_e[MTILE * 256];                    // eval state, f32
  __shared__ float s_k[MTILE * 256];                    // RK4 k accumulator, f32
  __shared__ _Float16 s_in[MTILE * LDA_IN] __attribute__((aligned(16)));
  __shared__ _Float16 s_h1[MTILE * LDH]    __attribute__((aligned(16)));
  __shared__ _Float16 s_h2[MTILE * LDH]    __attribute__((aligned(16)));

  const int tid  = threadIdx.x;
  const int lane = tid & 31;
  const int wave = tid >> 5;
  const int row0 = blockIdx.x * MTILE;

  gf16* w1f = (gf16*)wf;
  gf16* w2f = (gf16*)(wf + W1F_ELEMS);
  gf16* w3f = (gf16*)(wf + W1F_ELEMS + W2F_ELEMS);

  // Stage biases + initial state [x, 0].
  for (int i = tid; i < HIDN; i += NTHREADS) { sb1[i] = b1[i]; sb2[i] = b2[i]; }
  if (tid < DIMN) sb3[tid] = b3[tid];
  for (int i = tid; i < MTILE * DIMN; i += NTHREADS) {
    int r = i >> 7, c = i & 127;
    s_x[r * 256 + c]       = x[(size_t)(row0 + r) * DIMN + c];
    s_x[r * 256 + 128 + c] = 0.0f;
  }
  __syncthreads();

  const float dt = 0.1f;
#pragma unroll 1
  for (int step = 0; step < 10; ++step) {
    const float t0 = (float)step * 0.1f;
    // k-accumulator = 0, eval state = current state
    for (int i = tid; i < MTILE * 256; i += NTHREADS) { s_k[i] = 0.0f; s_e[i] = s_x[i]; }
    __syncthreads();

#pragma unroll 1
    for (int ev = 0; ev < 4; ++ev) {
      const float te = t0 + ((ev == 0) ? 0.0f : (ev == 3) ? dt : 0.5f * dt);
      const float wk = (ev == 1 || ev == 2) ? 2.0f : 1.0f;
      const float an = (ev == 2) ? dt : 0.5f * dt;  // step coeff for next eval state

      // Build padded f16 input [state | t | 0...] from eval state.
      for (int i = tid; i < MTILE * LDA_IN; i += NTHREADS) {
        int r = i / LDA_IN, c = i - r * LDA_IN;
        _Float16 v = (_Float16)0.0f;
        if (c < 256)       v = (_Float16)s_e[r * 256 + c];
        else if (c == 256) v = (_Float16)te;
        s_in[i] = v;
      }
      __syncthreads();

      mlp_layer<KT1>(s_in, LDA_IN, w1f, sb1, s_h1, LDH, wave, lane);  // layer 1
      __syncthreads();
      mlp_layer<KT2>(s_h1, LDH, w2f, sb2, s_h2, LDH, wave, lane);     // layer 2
      __syncthreads();

      // Layer 3: one 16x16 tile per wave (mt = wave>>3, nt = wave&7), no tanh.
      const int mt = wave >> 3, nt = wave & 7, m0 = mt << 4;
      const int col = lane & 15, hi = lane >> 4;
      v8f acc3 = splat8(sb3[(nt << 4) + col]);
      {
        v16h a0 = load_a(s_h2, LDH, m0, 0, lane);
        v16h b0 = load_b(opaq(w3f + ((size_t)nt << 9)), lane);
        v16h a1, b1v;
#pragma unroll 1
        for (int kk = 0; kk < KT2 / 2; ++kk) {
          const int k1 = 2 * kk + 1;
          a1  = load_a(s_h2, LDH, m0, k1 << 5, lane);
          b1v = load_b(opaq(w3f + ((size_t)(k1 * NT3 + nt) << 9)), lane);
          acc3 = wmma_f16(a0, b0, acc3);
          const int k2 = (2 * kk + 2 < KT2) ? 2 * kk + 2 : KT2 - 1;
          a0 = load_a(s_h2, LDH, m0, k2 << 5, lane);
          b0 = load_b(opaq(w3f + ((size_t)(k2 * NT3 + nt) << 9)), lane);
          acc3 = wmma_f16(a1, b1v, acc3);
        }
      }

      // Phase A: position half of k_i (velocity from eval state) + next eval pos.
      for (int i = tid; i < MTILE * DIMN; i += NTHREADS) {
        int r = i >> 7, c = i & 127;
        float v = s_e[r * 256 + 128 + c];
        s_k[r * 256 + c] += wk * v;
        if (ev != 3) s_e[r * 256 + c] = s_x[r * 256 + c] + an * v;
      }
      __syncthreads();

      // Phase B: acceleration half straight from WMMA accumulators.
#pragma unroll
      for (int j = 0; j < 8; ++j) {
        int m  = m0 + j + (hi << 3);
        int cc = 128 + (nt << 4) + col;
        float a = acc3[j];
        s_k[m * 256 + cc] += wk * a;
        if (ev != 3) s_e[m * 256 + cc] = s_x[m * 256 + cc] + an * a;
      }
      __syncthreads();
    }

    // state += dt/6 * (k1 + 2k2 + 2k3 + k4)
    for (int i = tid; i < MTILE * 256; i += NTHREADS) s_x[i] += (dt / 6.0f) * s_k[i];
    __syncthreads();
  }

  // Emit position part.
  for (int i = tid; i < MTILE * DIMN; i += NTHREADS) {
    int r = i >> 7, c = i & 127;
    out[(size_t)(row0 + r) * DIMN + c] = s_x[r * 256 + c];
  }
}

// ------------------------------ launcher -----------------------------------
extern "C" void kernel_launch(void* const* d_in, const int* in_sizes, int n_in,
                              void* d_out, int out_size, void* d_ws, size_t ws_size,
                              hipStream_t stream) {
  const float* x  = (const float*)d_in[0];
  const float* W1 = (const float*)d_in[1];
  const float* b1 = (const float*)d_in[2];
  const float* W2 = (const float*)d_in[3];
  const float* b2 = (const float*)d_in[4];
  const float* W3 = (const float*)d_in[5];
  const float* b3 = (const float*)d_in[6];
  _Float16* wf = (_Float16*)d_ws;          // 475136 halfs = ~0.93 MB scratch

  ode_prep_kernel<<<(WF_TOTAL + 255) / 256, 256, 0, stream>>>(W1, W2, W3, wf);
  ode_main_kernel<<<4096 / MTILE, NTHREADS, 0, stream>>>(x, b1, b2, b3, wf,
                                                         (float*)d_out);
}
